// MultiAgentsEncoder_48103633715685
// MI455X (gfx1250) — compile-verified, matrix operand order checked by reading, not run on hardware
//
#include <hip/hip_runtime.h>
#include <stdint.h>

// ---------------------------------------------------------------------------
// Problem constants (from reference)
// ---------------------------------------------------------------------------
static constexpr int T_   = 256;   // SRC_LEN
static constexpr int BT   = 32;    // B = BSZ * N_AGENTS
static constexpr int H    = 512;   // HSZ (== EMB)
static constexpr int G4   = 2048;  // 4*HSZ
static constexpr int NL   = 2;     // NLAYERS

typedef __bf16 bf16;
typedef __attribute__((ext_vector_type(16))) __bf16 v16bf;
typedef __attribute__((ext_vector_type(8)))  float  v8f;

// ---------------------------------------------------------------------------
// bf16 helpers (RNE convert; bit ops to avoid relying on fp conversions)
// ---------------------------------------------------------------------------
__device__ __forceinline__ float frombf(bf16 b) {
    uint32_t u = ((uint32_t)__builtin_bit_cast(unsigned short, b)) << 16;
    return __builtin_bit_cast(float, u);
}
__device__ __forceinline__ bf16 tobf(float f) {
    uint32_t u = __builtin_bit_cast(uint32_t, f);
    uint32_t r = u + 0x7FFFu + ((u >> 16) & 1u);
    return __builtin_bit_cast(bf16, (unsigned short)(r >> 16));
}
__device__ __forceinline__ float sigm(float x) {
    return 1.0f / (1.0f + __expf(-x));
}

// Load a 16-element bf16 WMMA fragment for one lane.
// p points at [row, baseK]; elems 0..7 = K baseK..baseK+7, elems 8..15 = K baseK+16..+23
__device__ __forceinline__ v16bf load_frag(const bf16* p) {
    union { uint4 u[2]; v16bf v; } f;
    f.u[0] = *reinterpret_cast<const uint4*>(p);
    f.u[1] = *reinterpret_cast<const uint4*>(p + 16);
    return f.v;
}

__device__ __forceinline__ v8f wmma_bf(v16bf a, v16bf b, v8f c) {
    return __builtin_amdgcn_wmma_f32_16x16x32_bf16(false, a, false, b, (short)0, c,
                                                   false, false);
}

// ---------------------------------------------------------------------------
// bf16 WMMA GEMM:  C[M,N] = A[M,K] * Bt[N,K]^T (+bias0+bias1)(+rowadd)(tanh)
// Each wave computes a 32x64 macro-tile (2 M-tiles x 4 N-tiles): per k-step it
// loads 2 A-frags + 4 B-frags and issues 8 WMMAs (register-level reuse), with
// software-pipelined (ping-pong) fragment loads so VMEM overlaps the WMMA pipe.
// block = 128 (4 waves, covering N256). grid = (N/256, M/32).
// Requires: M % 32 == 0, N % 256 == 0, K % 64 == 0.
// ---------------------------------------------------------------------------
__global__ __launch_bounds__(128)
void gemm_bf16_kernel(const bf16* __restrict__ A, const bf16* __restrict__ Bt,
                      const float* __restrict__ bias0, const float* __restrict__ bias1,
                      const float* __restrict__ rowadd,   // [32, N], indexed row&31
                      float* __restrict__ Cf, bf16* __restrict__ Cb,
                      int M, int N, int K, int dotanh)
{
    const int lane = threadIdx.x & 31;
    const int wave = threadIdx.x >> 5;
    const int n0   = (blockIdx.x * 4 + wave) * 64;
    const int m0   = blockIdx.y * 32;
    const int half = lane >> 4;
    const int lr   = lane & 15;

    const bf16* arow0 = A  + (size_t)(m0 + lr)      * K;
    const bf16* arow1 = A  + (size_t)(m0 + 16 + lr) * K;
    const bf16* brow0 = Bt + (size_t)(n0 + lr)      * K;
    const bf16* brow1 = Bt + (size_t)(n0 + 16 + lr) * K;
    const bf16* brow2 = Bt + (size_t)(n0 + 32 + lr) * K;
    const bf16* brow3 = Bt + (size_t)(n0 + 48 + lr) * K;

    v8f acc[2][4];
    for (int mt = 0; mt < 2; ++mt)
        for (int j = 0; j < 4; ++j) acc[mt][j] = (v8f){};

#define LOAD_SET(kk, X0, X1, Y0, Y1, Y2, Y3)                                   \
    do {                                                                       \
        const int bk_ = (kk) + 8 * half;                                       \
        X0 = load_frag(arow0 + bk_); X1 = load_frag(arow1 + bk_);              \
        Y0 = load_frag(brow0 + bk_); Y1 = load_frag(brow1 + bk_);              \
        Y2 = load_frag(brow2 + bk_); Y3 = load_frag(brow3 + bk_);              \
    } while (0)

#define WMMA8(X0, X1, Y0, Y1, Y2, Y3)                                          \
    do {                                                                       \
        acc[0][0] = wmma_bf(X0, Y0, acc[0][0]);                                \
        acc[0][1] = wmma_bf(X0, Y1, acc[0][1]);                                \
        acc[0][2] = wmma_bf(X0, Y2, acc[0][2]);                                \
        acc[0][3] = wmma_bf(X0, Y3, acc[0][3]);                                \
        acc[1][0] = wmma_bf(X1, Y0, acc[1][0]);                                \
        acc[1][1] = wmma_bf(X1, Y1, acc[1][1]);                                \
        acc[1][2] = wmma_bf(X1, Y2, acc[1][2]);                                \
        acc[1][3] = wmma_bf(X1, Y3, acc[1][3]);                                \
    } while (0)

    v16bf a0, a1, b0, b1, b2, b3;        // stage 0 buffers
    v16bf A0, A1, B0, B1, B2, B3;        // stage 1 buffers

    LOAD_SET(0, a0, a1, b0, b1, b2, b3);
    for (int k0 = 0; k0 < K; k0 += 64) {
        LOAD_SET(k0 + 32, A0, A1, B0, B1, B2, B3);
        WMMA8(a0, a1, b0, b1, b2, b3);
        if (k0 + 64 < K) LOAD_SET(k0 + 64, a0, a1, b0, b1, b2, b3);
        WMMA8(A0, A1, B0, B1, B2, B3);
    }
#undef LOAD_SET
#undef WMMA8

    for (int j = 0; j < 4; ++j) {
        const int col = n0 + j * 16 + lr;
        float badd = 0.0f;
        if (bias0) badd += bias0[col];
        if (bias1) badd += bias1[col];
        for (int mt = 0; mt < 2; ++mt) {
            for (int v = 0; v < 8; ++v) {
                const int row = m0 + mt * 16 + 8 * half + v;
                float x = acc[mt][j][v] + badd;
                if (rowadd) x += rowadd[(size_t)(row & 31) * N + col];
                if (dotanh) x = tanhf(x);
                if (Cf) Cf[(size_t)row * N + col] = x;
                if (Cb) Cb[(size_t)row * N + col] = tobf(x);
            }
        }
    }
}

// ---------------------------------------------------------------------------
// Sequential LSTM over T steps, one direction per block (grid = 2).
// 1024 threads = 32 waves; wave w owns hidden-unit columns [w*16, w*16+16)
// for ALL FOUR gates and both 16-row M-tiles, so the gate nonlinearity needs
// no cross-wave exchange. h lives in LDS (bf16, 32KB); c lives in registers.
// Whh (bf16, [2048,512] row-major) streams from L2 each step.
// xb = x@Wih^T + bih + bhh precomputed as bf16 [T,32,2048].
// (No pipeline buffers here: 8 waves/SIMD * ~120 VGPRs already fills the RF.)
// ---------------------------------------------------------------------------
__global__ __launch_bounds__(1024)
void lstm_seq_kernel(const bf16* __restrict__ xb0, const bf16* __restrict__ xb1,
                     const bf16* __restrict__ whh0, const bf16* __restrict__ whh1,
                     bf16* __restrict__ y0, bf16* __restrict__ y1,
                     float* __restrict__ ct0, float* __restrict__ ct1)
{
    const int dir = blockIdx.x;
    const bf16* xb  = dir ? xb1  : xb0;
    const bf16* whh = dir ? whh1 : whh0;
    bf16*  y  = dir ? y1  : y0;
    float* ct = dir ? ct1 : ct0;

    __shared__ bf16 hsh[BT * H];          // 32 KB: current hidden state (bf16)

    const int tid = threadIdx.x;
    // zero h state
    uint32_t* hz = reinterpret_cast<uint32_t*>(hsh);
    for (int i = tid; i < BT * H / 2; i += 1024) hz[i] = 0u;
    __syncthreads();

    const int lane = tid & 31;
    const int w    = tid >> 5;
    const int half = lane >> 4;
    const int lr   = lane & 15;
    const int jc   = w * 16 + lr;         // hidden-unit column this lane owns

    float c[2][8];
    for (int mt = 0; mt < 2; ++mt)
        for (int v = 0; v < 8; ++v) c[mt][v] = 0.0f;

    for (int t = 0; t < T_; ++t) {
        const int tt = dir ? (T_ - 1 - t) : t;

        // init accumulators with xb[tt] (input projection + biases)
        v8f acc[2][4];
        const bf16* xbt = xb + (size_t)tt * BT * G4;
        for (int mt = 0; mt < 2; ++mt)
            for (int g = 0; g < 4; ++g) {
                v8f a;
                for (int v = 0; v < 8; ++v)
                    a[v] = frombf(xbt[(size_t)(mt * 16 + 8 * half + v) * G4 +
                                      g * 512 + jc]);
                acc[mt][g] = a;
            }

        // h @ Whh^T : K = 512, k-step 32 per WMMA; 8 output tiles per wave
        for (int k0 = 0; k0 < H; k0 += 32) {
            const int bk = k0 + 8 * half;
            v16bf a0 = load_frag(hsh + (size_t)(lr)      * H + bk);
            v16bf a1 = load_frag(hsh + (size_t)(16 + lr) * H + bk);
            for (int g = 0; g < 4; ++g) {
                v16bf b = load_frag(whh + (size_t)(g * 512 + w * 16 + lr) * H + bk);
                acc[0][g] = wmma_bf(a0, b, acc[0][g]);
                acc[1][g] = wmma_bf(a1, b, acc[1][g]);
            }
        }

        __syncthreads();   // all waves done reading hsh before overwriting

        for (int mt = 0; mt < 2; ++mt) {
            for (int v = 0; v < 8; ++v) {
                const float iv = sigm(acc[mt][0][v]);
                const float fv = sigm(acc[mt][1][v]);
                const float gv = tanhf(acc[mt][2][v]);
                const float ov = sigm(acc[mt][3][v]);
                const float cn = fv * c[mt][v] + iv * gv;
                c[mt][v] = cn;
                const float hv = ov * tanhf(cn);
                const int row = mt * 16 + 8 * half + v;
                const bf16 hb = tobf(hv);
                hsh[row * H + jc] = hb;
                y[((size_t)tt * BT + row) * H + jc] = hb;
            }
        }
        __syncthreads();   // h fully written before next step's reads
    }

    // final cell state (f32)
    for (int mt = 0; mt < 2; ++mt)
        for (int v = 0; v < 8; ++v) {
            const int row = mt * 16 + 8 * half + v;
            ct[(size_t)row * H + jc] = c[mt][v];
        }
}

// ---------------------------------------------------------------------------
// Small elementwise kernels
// ---------------------------------------------------------------------------
__global__ void f2bf_kernel(const float* __restrict__ in, bf16* __restrict__ out,
                            size_t n)
{
    size_t i = (size_t)blockIdx.x * blockDim.x + threadIdx.x;
    if (i < n) out[i] = tobf(in[i]);
}

// enccat[m, 0:512] = yf[m], enccat[m, 512:1024] = yb[m]
__global__ void concat_kernel(const bf16* __restrict__ yf, const bf16* __restrict__ yb,
                              bf16* __restrict__ out, size_t rows)
{
    size_t i = (size_t)blockIdx.x * blockDim.x + threadIdx.x;
    if (i >= rows * 1024) return;
    size_t r = i >> 10;
    int k = (int)(i & 1023);
    out[i] = (k < 512) ? yf[r * 512 + k] : yb[r * 512 + (k - 512)];
}

// hcat[b] = [ yf[T-1, b], yb[0, b] ]   (final hidden of fwd / bwd scans)
__global__ void hcat_kernel(const bf16* __restrict__ yf, const bf16* __restrict__ yb,
                            bf16* __restrict__ out)
{
    int i = blockIdx.x * blockDim.x + threadIdx.x;
    if (i >= BT * 1024) return;
    int b = i >> 10, k = i & 1023;
    out[i] = (k < 512) ? yf[((size_t)(T_ - 1) * BT + b) * 512 + k]
                       : yb[(size_t)b * 512 + (k - 512)];
}

// leave-one-out mean over agents (groups of 4 rows), output bf16
__global__ void msg_kernel(const float* __restrict__ hproj, bf16* __restrict__ msg)
{
    int i = blockIdx.x * blockDim.x + threadIdx.x;
    if (i >= BT * 512) return;
    int b = i >> 9, j = i & 511;
    int gi = b >> 2;
    float s = 0.0f;
    for (int a = 0; a < 4; ++a) s += hproj[(size_t)((gi << 2) + a) * 512 + j];
    msg[i] = tobf((s - hproj[(size_t)b * 512 + j]) * (1.0f / 3.0f));
}

// h1 = hproj rows {0,4,...,28}; c1 = [ctf | ctb] rows {0,4,...,28}
__global__ void tail_kernel(const float* __restrict__ hproj,
                            const float* __restrict__ ctf,
                            const float* __restrict__ ctb,
                            float* __restrict__ out_h1, float* __restrict__ out_c1)
{
    int i = blockIdx.x * blockDim.x + threadIdx.x;
    if (i < 8 * 512) {
        int b = i >> 9, j = i & 511;
        out_h1[i] = hproj[(size_t)(b * 4) * 512 + j];
    } else if (i < 8 * 512 + 8 * 1024) {
        int k = i - 4096;
        int b = k >> 10, j = k & 1023;
        out_c1[(size_t)b * 1024 + j] =
            (j < 512) ? ctf[(size_t)(b * 4) * 512 + j]
                      : ctb[(size_t)(b * 4) * 512 + (j - 512)];
    }
}

// ---------------------------------------------------------------------------
// Host-side orchestration
// ---------------------------------------------------------------------------
static inline void launch_gemm(const void* A, const void* Bt,
                               const float* b0, const float* b1, const float* rowadd,
                               float* Cf, void* Cb,
                               int M, int N, int K, int dotanh, hipStream_t s)
{
    dim3 grid(N / 256, M / 32);
    gemm_bf16_kernel<<<grid, 128, 0, s>>>((const bf16*)A, (const bf16*)Bt,
                                          b0, b1, rowadd, Cf, (bf16*)Cb,
                                          M, N, K, dotanh);
}

static inline void launch_f2bf(const float* in, void* out, size_t n, hipStream_t s)
{
    f2bf_kernel<<<(unsigned)((n + 255) / 256), 256, 0, s>>>(in, (bf16*)out, n);
}

extern "C" void kernel_launch(void* const* d_in, const int* in_sizes, int n_in,
                              void* d_out, int out_size, void* d_ws, size_t ws_size,
                              hipStream_t stream)
{
    (void)in_sizes; (void)n_in; (void)out_size; (void)ws_size;
    const float* x    = (const float*)d_in[0];
    const float* lWf  = (const float*)d_in[1];
    const float* lUf  = (const float*)d_in[2];
    const float* lbf  = (const float*)d_in[3];
    const float* lbbf = (const float*)d_in[4];
    const float* lWb  = (const float*)d_in[5];
    const float* lUb  = (const float*)d_in[6];
    const float* lbb  = (const float*)d_in[7];
    const float* lbbb = (const float*)d_in[8];
    const float* Wp   = (const float*)d_in[9];
    const float* bp   = (const float*)d_in[10];
    const float* Wmsg = (const float*)d_in[11];
    const float* Wseq = (const float*)d_in[12];
    const float* cWf  = (const float*)d_in[13];
    const float* cUf  = (const float*)d_in[14];
    const float* cbf  = (const float*)d_in[15];
    const float* cbbf = (const float*)d_in[16];
    const float* cWb  = (const float*)d_in[17];
    const float* cUb  = (const float*)d_in[18];
    const float* cbb  = (const float*)d_in[19];
    const float* cbbb = (const float*)d_in[20];

    const size_t MR = (size_t)T_ * BT;          // 8192 rows

    // ---- workspace layout ----
    char* ws = (char*)d_ws;
    size_t off = 0;
    auto alloc = [&](size_t bytes) -> char* {
        char* p = ws + off;
        off = (off + bytes + 255) & ~(size_t)255;
        return p;
    };
    bf16*  XBF   = (bf16*)alloc(MR * H * 2);                 // x in bf16
    bf16*  LWF   = (bf16*)alloc((size_t)G4 * H * 2);
    bf16*  LUF   = (bf16*)alloc((size_t)G4 * H * 2);
    bf16*  LWB   = (bf16*)alloc((size_t)G4 * H * 2);
    bf16*  LUB   = (bf16*)alloc((size_t)G4 * H * 2);
    bf16*  CWF   = (bf16*)alloc((size_t)NL * G4 * H * 2);
    bf16*  CUF   = (bf16*)alloc((size_t)NL * G4 * H * 2);
    bf16*  CWB   = (bf16*)alloc((size_t)NL * G4 * H * 2);
    bf16*  CUB   = (bf16*)alloc((size_t)NL * G4 * H * 2);
    bf16*  WPB   = (bf16*)alloc((size_t)H * 2 * H * 2);      // [512,1024]
    bf16*  WMSGB = (bf16*)alloc((size_t)H * H * 2);
    bf16*  WSEQB = (bf16*)alloc((size_t)H * H * 2);
    bf16*  XBFW  = (bf16*)alloc(MR * G4 * 2);                // xb fwd (bf16)
    bf16*  XBBW  = (bf16*)alloc(MR * G4 * 2);                // xb bwd
    bf16*  YF    = (bf16*)alloc(MR * H * 2);
    bf16*  YB    = (bf16*)alloc(MR * H * 2);
    float* CTF   = (float*)alloc((size_t)BT * H * 4);
    float* CTB   = (float*)alloc((size_t)BT * H * 4);
    bf16*  ENCC  = (bf16*)alloc(MR * 2 * H * 2);             // concat [8192,1024]
    float* ENC   = (float*)alloc(MR * H * 4);
    bf16*  ENCB  = (bf16*)alloc(MR * H * 2);
    bf16*  NXT   = (bf16*)alloc(MR * H * 2);
    bf16*  HCAT  = (bf16*)alloc((size_t)BT * 2 * H * 2);
    float* HPROJ = (float*)alloc((size_t)BT * H * 4);
    bf16*  MSGB  = (bf16*)alloc((size_t)BT * H * 2);
    float* MSGP  = (float*)alloc((size_t)BT * H * 4);

    // ---- 1. convert inputs/weights to bf16 ----
    launch_f2bf(x,    XBF,   MR * H, stream);
    launch_f2bf(lWf,  LWF,   (size_t)G4 * H, stream);
    launch_f2bf(lUf,  LUF,   (size_t)G4 * H, stream);
    launch_f2bf(lWb,  LWB,   (size_t)G4 * H, stream);
    launch_f2bf(lUb,  LUB,   (size_t)G4 * H, stream);
    launch_f2bf(cWf,  CWF,   (size_t)NL * G4 * H, stream);
    launch_f2bf(cUf,  CUF,   (size_t)NL * G4 * H, stream);
    launch_f2bf(cWb,  CWB,   (size_t)NL * G4 * H, stream);
    launch_f2bf(cUb,  CUB,   (size_t)NL * G4 * H, stream);
    launch_f2bf(Wp,   WPB,   (size_t)H * 2 * H, stream);
    launch_f2bf(Wmsg, WMSGB, (size_t)H * H, stream);
    launch_f2bf(Wseq, WSEQB, (size_t)H * H, stream);

    // ---- 2. local layer: input projections (parallel over T) ----
    launch_gemm(XBF, LWF, lbf, lbbf, nullptr, nullptr, XBFW, (int)MR, G4, H, 0, stream);
    launch_gemm(XBF, LWB, lbb, lbbb, nullptr, nullptr, XBBW, (int)MR, G4, H, 0, stream);

    // ---- 3. local layer: sequential recurrence (fwd+bwd in parallel) ----
    lstm_seq_kernel<<<2, 1024, 0, stream>>>(XBFW, XBBW, LUF, LUB, YF, YB, CTF, CTB);

    // ---- 4. concat + projection Wp ----
    concat_kernel<<<(unsigned)((MR * 1024 + 255) / 256), 256, 0, stream>>>(YF, YB, ENCC, MR);
    launch_gemm(ENCC, WPB, bp, nullptr, nullptr, ENC, ENCB, (int)MR, H, 2 * H, 0, stream);
    hcat_kernel<<<(BT * 1024 + 255) / 256, 256, 0, stream>>>(YF, YB, HCAT);
    launch_gemm(HCAT, WPB, bp, nullptr, nullptr, HPROJ, nullptr, BT, H, 2 * H, 0, stream);

    // ---- 5. communication layers ----
    for (int l = 0; l < NL; ++l) {
        const size_t wofs = (size_t)l * G4 * H;
        msg_kernel<<<(BT * 512 + 255) / 256, 256, 0, stream>>>(HPROJ, MSGB);
        launch_gemm(MSGB, WMSGB, nullptr, nullptr, nullptr, MSGP, nullptr, BT, H, H, 0, stream);
        // nxt = tanh(msg@Wmsg^T (broadcast over t) + enc@Wseq^T)
        launch_gemm(ENCB, WSEQB, nullptr, nullptr, MSGP, nullptr, NXT, (int)MR, H, H, 1, stream);
        // input projections for this layer's BiLSTM
        launch_gemm(NXT, CWF + wofs, cbf + l * G4, cbbf + l * G4, nullptr,
                    nullptr, XBFW, (int)MR, G4, H, 0, stream);
        launch_gemm(NXT, CWB + wofs, cbb + l * G4, cbbb + l * G4, nullptr,
                    nullptr, XBBW, (int)MR, G4, H, 0, stream);
        lstm_seq_kernel<<<2, 1024, 0, stream>>>(XBFW, XBBW, CUF + wofs, CUB + wofs,
                                                YF, YB, CTF, CTB);
        concat_kernel<<<(unsigned)((MR * 1024 + 255) / 256), 256, 0, stream>>>(YF, YB, ENCC, MR);
        launch_gemm(ENCC, WPB, bp, nullptr, nullptr, ENC, ENCB, (int)MR, H, 2 * H, 0, stream);
        hcat_kernel<<<(BT * 1024 + 255) / 256, 256, 0, stream>>>(YF, YB, HCAT);
        launch_gemm(HCAT, WPB, bp, nullptr, nullptr, HPROJ, nullptr, BT, H, 2 * H, 0, stream);
    }

    // ---- 6. outputs: enc4 (direct layout match), h1, c1 ----
    float* out = (float*)d_out;
    hipMemcpyAsync(out, ENC, MR * H * sizeof(float), hipMemcpyDeviceToDevice, stream);
    tail_kernel<<<(8 * 512 + 8 * 1024 + 255) / 256, 256, 0, stream>>>(
        HPROJ, CTF, CTB, out + MR * H, out + MR * H + 8 * 512);
}
